// MFN_55954833932981
// MI455X (gfx1250) — matrix operand order repaired and checked by reading, not compiled, not run on hardware
//
#include <hip/hip_runtime.h>

typedef __bf16 bf16;
typedef __attribute__((ext_vector_type(16))) __bf16 v16bf;
typedef __attribute__((ext_vector_type(8)))  __bf16 v8bf;
typedef __attribute__((ext_vector_type(8)))  float  v8f;

#define WMMA_BF16(a, b, c) \
  __builtin_amdgcn_wmma_f32_16x16x32_bf16(false, (a), false, (b), (short)0, (c), false, false)

// ---------------- problem constants ----------------
constexpr int T_STEPS = 64;
constexpr int BATCH   = 4096;
constexpr int ROWS    = 16;     // batch rows per block (one WMMA M tile)
constexpr int THREADS = 256;    // 8 waves (wave32)

// ---------------- packed weight offsets (bf16 elements) ----------------
// layout per matrix: [dinP/32 ktiles][dout/16 ntiles][32 lanes][16 elems]
constexpr long OFF_WIH_L  = 0;
constexpr long OFF_WHH_L  = OFF_WIH_L  + 320L  * 1024;
constexpr long OFF_WIH_A  = OFF_WHH_L  + 256L  * 1024;
constexpr long OFF_WHH_A  = OFF_WIH_A  + 128L  * 512;
constexpr long OFF_WIH_V  = OFF_WHH_A  + 128L  * 512;
constexpr long OFF_WHH_V  = OFF_WIH_V  + 128L  * 512;
constexpr long OFF_ATT1W1 = OFF_WHH_V  + 128L  * 512;
constexpr long OFF_ATT1W2 = OFF_ATT1W1 + 1024L * 512;
constexpr long OFF_ATT2W1 = OFF_ATT1W2 + 512L  * 1024;
constexpr long OFF_ATT2W2 = OFF_ATT2W1 + 1024L * 512;
constexpr long OFF_G1W1   = OFF_ATT2W2 + 512L  * 512;
constexpr long OFF_G1W2   = OFF_G1W1   + 1536L * 512;
constexpr long OFF_G2W1   = OFF_G1W2   + 512L  * 512;
constexpr long OFF_G2W2   = OFF_G2W1   + 1536L * 512;
constexpr long OFF_OUTW1  = OFF_G2W2   + 512L  * 512;
constexpr long WS_ELEMS   = OFF_OUTW1  + 1024L * 512;

// ---------------- LDS layout (bytes) ----------------
// X region [0,18432): XL 16x320 bf16 | XA 16x128 | XV 16x128
//   aliases: LOGHI (logit cols 512..1023, 16x512 bf16) and CHAT (16x512 bf16)
constexpr int OFF_XL    = 0;
constexpr int OFF_XA    = 10240;
constexpr int OFF_XV    = 14336;
constexpr int OFF_H     = 18432;   // h state  16x512 bf16 (l|a|v)
constexpr int OFF_CSTAR = 34816;   // cStar / attended 16x1024 bf16
constexpr int OFF_HID0  = 67584;   // att1h / h-staging / softmax-scratch / g1h / out-hid
constexpr int OFF_HID1  = 83968;   // logit cols 0..511 / att2h / g2h / f32 scratch
constexpr int OFF_MEM   = 100352;  // mem state 16x512 f32
constexpr int SMEM_BYTES = 133120;

// ---------------- device helpers ----------------
__device__ __forceinline__ float sigmoid_f(float x) {
  return 1.f / (1.f + __expf(-x));
}
__device__ __forceinline__ float tanh_f(float x) {
  x = fminf(fmaxf(x, -15.f), 15.f);
  float e = __expf(2.f * x);
  return (e - 1.f) / (e + 1.f);
}

// A fragment (16x32 bf16, MxK) from row-major bf16 LDS.
// Lanes 0-15: K 0..7 then 16..23 ; lanes 16-31: K 8..15 then 24..31 (ISA 7.12.2)
__device__ __forceinline__ v16bf load_a_bf16(const bf16* base, int stride, int k0, int lane) {
  int m = lane & 15, half = lane >> 4;
  const bf16* p = base + m * stride + k0 + half * 8;
  v8bf lo = *(const v8bf*)p;
  v8bf hi = *(const v8bf*)(p + 16);
  v16bf a;
#pragma unroll
  for (int e = 0; e < 8; ++e) { a[e] = lo[e]; a[8 + e] = hi[e]; }
  return a;
}

// same but source is f32 LDS (converted on the fly)
__device__ __forceinline__ v16bf load_a_f32(const float* base, int stride, int k0, int lane) {
  int m = lane & 15, half = lane >> 4;
  const float* p = base + m * stride + k0 + half * 8;
  v8f lo = *(const v8f*)p;
  v8f hi = *(const v8f*)(p + 16);
  v16bf a;
#pragma unroll
  for (int e = 0; e < 8; ++e) { a[e] = (bf16)lo[e]; a[8 + e] = (bf16)hi[e]; }
  return a;
}

// 4 adjacent output tiles sharing the A fragment (amortizes LDS A traffic 4x)
template <typename LA>
__device__ __forceinline__ void gemm4(v8f acc[4], LA&& loadA, const bf16* __restrict__ wp,
                                      int ntiles, int nt0, int ktn, int lane) {
  for (int kt = 0; kt < ktn; ++kt) {
    v16bf a = loadA(kt);
#pragma unroll
    for (int i = 0; i < 4; ++i) {
      v16bf b = ((const v16bf*)(wp + (((long)kt * ntiles + (nt0 + i)) << 9)))[lane];
      acc[i] = WMMA_BF16(a, b, acc[i]);
    }
  }
}

// store one 16x16 f32 acc tile to bf16 LDS with bias+activation (0=none,1=relu,2=tanh)
__device__ __forceinline__ void store_tile(bf16* dst, int stride, int c0, const v8f& acc,
                                           float b, int act, int mrow, int ncol) {
#pragma unroll
  for (int r = 0; r < 8; ++r) {
    float v = acc[r] + b;
    if (act == 1) v = fmaxf(v, 0.f);
    else if (act == 2) v = tanh_f(v);
    dst[(mrow + r) * stride + c0 + ncol] = (bf16)v;
  }
}

// ---------------- weight packer ----------------
// src: row-major f32 (dout, din). dst tile layout documented above; K padded to dinP, zero fill.
__global__ void pack_weights(const float* __restrict__ src, bf16* __restrict__ dst,
                             int dout, int din, int dinP) {
  int ktiles = dinP >> 5, ntiles = dout >> 4;
  long total = (long)ktiles * ntiles * 512;
  for (long i = (long)blockIdx.x * blockDim.x + threadIdx.x; i < total;
       i += (long)gridDim.x * blockDim.x) {
    int e = (int)(i & 15);
    int lane = (int)((i >> 4) & 31);
    long tile = i >> 9;
    int nt = (int)(tile % ntiles);
    int kt = (int)(tile / ntiles);
    int n = nt * 16 + (lane & 15);
    int k = kt * 32 + ((lane >> 4) * 8) + (e & 7) + ((e >> 3) * 16);
    float v = (k < din) ? src[(long)n * din + k] : 0.f;
    dst[i] = (bf16)v;
  }
}

// ---------------- fused MFN kernel: 16 batch rows per block, all 64 steps ----------------
__global__ void __launch_bounds__(THREADS)
mfn_main(const int* __restrict__ x_idx, const float* __restrict__ x_av,
         const float* __restrict__ c_l0, const float* __restrict__ c_a0,
         const float* __restrict__ c_v0, const float* __restrict__ mem0,
         const float* __restrict__ embed,
         const float* __restrict__ bih_l, const float* __restrict__ bhh_l,
         const float* __restrict__ bih_a, const float* __restrict__ bhh_a,
         const float* __restrict__ bih_v, const float* __restrict__ bhh_v,
         const float* __restrict__ att1_b1, const float* __restrict__ att1_b2,
         const float* __restrict__ att2_b1, const float* __restrict__ att2_b2,
         const float* __restrict__ g1_b1, const float* __restrict__ g1_b2,
         const float* __restrict__ g2_b1, const float* __restrict__ g2_b2,
         const float* __restrict__ out_b1, const float* __restrict__ out_w2,
         const float* __restrict__ out_b2,
         const bf16* __restrict__ wpack, float* __restrict__ out) {
  extern __shared__ char smem[];
  bf16*  XL    = (bf16*)(smem + OFF_XL);
  bf16*  XA    = (bf16*)(smem + OFF_XA);
  bf16*  XV    = (bf16*)(smem + OFF_XV);
  bf16*  LOGHI = (bf16*)(smem + OFF_XL);   // alias (phase C/D)
  bf16*  CHAT  = (bf16*)(smem + OFF_XL);   // alias (phase F..H)
  bf16*  H     = (bf16*)(smem + OFF_H);
  bf16*  CSTAR = (bf16*)(smem + OFF_CSTAR);
  bf16*  HID0  = (bf16*)(smem + OFF_HID0);
  bf16*  HTMP  = (bf16*)(smem + OFF_HID0); // alias (phase A h-staging)
  bf16*  HID1  = (bf16*)(smem + OFF_HID1);
  float* MEMF  = (float*)(smem + OFF_MEM);
  float* RED   = (float*)(smem + OFF_HID0); // alias (softmax reduction)

  const int tid  = threadIdx.x;
  const int wave = tid >> 5;
  const int lane = tid & 31;
  const int row0 = blockIdx.x * ROWS;
  const int mrow = (lane >> 4) * 8;  // D-tile row base for this lane
  const int ncol = lane & 15;        // D-tile column for this lane

  // ---- init: h=0, mem from input, LSTM c state in registers ----
  for (int i = tid; i < 16 * 512; i += THREADS) H[i] = (bf16)0.f;
  for (int i = tid; i < 16 * 512; i += THREADS) {
    int m = i >> 9, c = i & 511;
    MEMF[i] = mem0[(long)(row0 + m) * 512 + c];
  }
  float creg[4][8];  // per-wave persistent cell state (4 gate-groups x 8 rows)
#pragma unroll
  for (int gi = 0; gi < 4; ++gi) {
    int grp = wave * 4 + gi;
    const float* cin; int width, j;
    if (grp < 16)      { cin = c_l0; width = 256; j = grp; }
    else if (grp < 24) { cin = c_a0; width = 128; j = grp - 16; }
    else               { cin = c_v0; width = 128; j = grp - 24; }
#pragma unroll
    for (int r = 0; r < 8; ++r)
      creg[gi][r] = cin[(long)(row0 + mrow + r) * width + j * 16 + ncol];
  }
  __syncthreads();

  // Opaque zero offset, re-laundered every timestep. Adding it to the weight/bias
  // base pointers makes every weight address formally loop-variant (defeats LICM
  // hoisting of 64x-invariant fragments, which previously caused scratch spills)
  // while keeping pointer provenance intact so the loads lower to global_load_b128
  // (LOADcnt only) rather than flat_load (LOADcnt+DScnt coupled waits).
  long lz = 0;

#pragma clang loop unroll(disable)
  for (int t = 0; t < T_STEPS; ++t) {
    asm volatile("" : "+s"(lz));
    const bf16*  wp   = wpack + lz;
    const float* Bihl = bih_l + lz;  const float* Bhhl = bhh_l + lz;
    const float* Biha = bih_a + lz;  const float* Bhha = bhh_a + lz;
    const float* Bihv = bih_v + lz;  const float* Bhhv = bhh_v + lz;
    const float* A1b1 = att1_b1 + lz; const float* A1b2 = att1_b2 + lz;
    const float* A2b1 = att2_b1 + lz; const float* A2b2 = att2_b2 + lz;
    const float* G1b1 = g1_b1 + lz;  const float* G1b2 = g1_b2 + lz;
    const float* G2b1 = g2_b1 + lz;  const float* G2b2 = g2_b2 + lz;

    // ---- Phase X: stage inputs (embed gather -> bf16, split x_av) ----
    for (int i = tid; i < 16 * 320; i += THREADS) {
      int m = i / 320, k = i - m * 320;
      int idx = x_idx[(long)t * BATCH + row0 + m];
      XL[i] = (bf16)((k < 300) ? embed[(long)idx * 300 + k] : 0.f);
    }
    for (int i = tid; i < 16 * 256; i += THREADS) {
      int m = i >> 8, k = i & 255;
      float v = x_av[((long)t * BATCH + row0 + m) * 256 + k];
      if (k < 128) XA[m * 128 + k] = (bf16)v;
      else         XV[m * 128 + (k - 128)] = (bf16)v;
    }
    __syncthreads();

    // ---- Phase A: three LSTMs (i/f/g/o tiles share A fragments) ----
#pragma unroll
    for (int gi = 0; gi < 4; ++gi) {
      int grp = wave * 4 + gi;
      const bf16 *wih, *whh, *xsrc;
      const float *bip, *bhp;
      int j, dh, cb, xkt, xstride;
      if (grp < 16) {
        j = grp; dh = 256; cb = 0;
        wih = wp + OFF_WIH_L; whh = wp + OFF_WHH_L;
        xsrc = XL; xkt = 10; xstride = 320; bip = Bihl; bhp = Bhhl;
      } else if (grp < 24) {
        j = grp - 16; dh = 128; cb = 256;
        wih = wp + OFF_WIH_A; whh = wp + OFF_WHH_A;
        xsrc = XA; xkt = 4; xstride = 128; bip = Biha; bhp = Bhha;
      } else {
        j = grp - 24; dh = 128; cb = 384;
        wih = wp + OFF_WIH_V; whh = wp + OFF_WHH_V;
        xsrc = XV; xkt = 4; xstride = 128; bip = Bihv; bhp = Bhhv;
      }
      int ntg = dh >> 4, ntiles = dh >> 2;  // tiles/gate, total tiles (4*dh/16)
      v8f acc[4] = {};
      for (int kt = 0; kt < xkt; ++kt) {
        v16bf a = load_a_bf16(xsrc, xstride, kt * 32, lane);
#pragma unroll
        for (int q = 0; q < 4; ++q) {
          v16bf b = ((const v16bf*)(wih + (((long)kt * ntiles + q * ntg + j) << 9)))[lane];
          acc[q] = WMMA_BF16(a, b, acc[q]);
        }
      }
      int hkt = dh >> 5;
      for (int kt = 0; kt < hkt; ++kt) {
        v16bf a = load_a_bf16(H + cb, 512, kt * 32, lane);
#pragma unroll
        for (int q = 0; q < 4; ++q) {
          v16bf b = ((const v16bf*)(whh + (((long)kt * ntiles + q * ntg + j) << 9)))[lane];
          acc[q] = WMMA_BF16(a, b, acc[q]);
        }
      }
      float bv[4];
#pragma unroll
      for (int q = 0; q < 4; ++q)
        bv[q] = bip[q * dh + j * 16 + ncol] + bhp[q * dh + j * 16 + ncol];
      int fcol = cb + j * 16 + ncol;
#pragma unroll
      for (int r = 0; r < 8; ++r) {
        int m = mrow + r;
        float iv = sigmoid_f(acc[0][r] + bv[0]);
        float fv = sigmoid_f(acc[1][r] + bv[1]);
        float gv = tanh_f(acc[2][r] + bv[2]);
        float ov = sigmoid_f(acc[3][r] + bv[3]);
        float cold = creg[gi][r];
        float cnew = fv * cold + iv * gv;
        float hnew = ov * tanh_f(cnew);
        CSTAR[m * 1024 + fcol]       = (bf16)cold;  // old c (first half of cStar)
        CSTAR[m * 1024 + 512 + fcol] = (bf16)cnew;  // new c (second half)
        HTMP[m * 512 + fcol]         = (bf16)hnew;  // staged h'
        creg[gi][r] = cnew;
      }
    }
    __syncthreads();
    for (int i = tid; i < 16 * 512; i += THREADS) H[i] = HTMP[i];  // commit h'
    __syncthreads();

    // ---- Phase B: att1 hidden = relu(cStar @ w1^T + b1) -> HID0 ----
    {
      v8f acc[4] = {};
      int nt0 = wave * 4;
      auto lA = [&](int kt) { return load_a_bf16(CSTAR, 1024, kt * 32, lane); };
      gemm4(acc, lA, wp + OFF_ATT1W1, 32, nt0, 32, lane);
#pragma unroll
      for (int i = 0; i < 4; ++i)
        store_tile(HID0, 512, (nt0 + i) * 16, acc[i],
                   A1b1[(nt0 + i) * 16 + ncol], 1, mrow, ncol);
    }
    __syncthreads();

    // ---- Phase C: logits = HID0 @ w2^T + b2 -> HID1 | LOGHI ----
    for (int p = 0; p < 2; ++p) {
      v8f acc[4] = {};
      int nt0 = wave * 8 + p * 4;
      auto lA = [&](int kt) { return load_a_bf16(HID0, 512, kt * 32, lane); };
      gemm4(acc, lA, wp + OFF_ATT1W2, 64, nt0, 16, lane);
      bf16* dstb = (nt0 < 32) ? HID1 : LOGHI;
      int local0 = (nt0 < 32) ? nt0 : nt0 - 32;
#pragma unroll
      for (int i = 0; i < 4; ++i)
        store_tile(dstb, 512, (local0 + i) * 16, acc[i],
                   A1b2[(nt0 + i) * 16 + ncol], 0, mrow, ncol);
    }
    __syncthreads();

    // ---- Phase D: softmax over 1024 feats/row, attended = p * cStar (in place) ----
    {
      int m = tid >> 4, chunk = tid & 15;
      const bf16* lsrc = (chunk < 8) ? (HID1 + m * 512 + chunk * 64)
                                     : (LOGHI + m * 512 + (chunk - 8) * 64);
      float mx = -1e30f;
      for (int k = 0; k < 64; ++k) mx = fmaxf(mx, (float)lsrc[k]);
      RED[tid] = mx;
      __syncthreads();
      if (tid < 16) {
        float v = RED[tid * 16];
        for (int k = 1; k < 16; ++k) v = fmaxf(v, RED[tid * 16 + k]);
        RED[256 + tid] = v;
      }
      __syncthreads();
      float rmax = RED[256 + m];
      float s = 0.f;
      for (int k = 0; k < 64; ++k) s += __expf((float)lsrc[k] - rmax);
      RED[tid] = s;
      __syncthreads();
      if (tid < 16) {
        float v = 0.f;
        for (int k = 0; k < 16; ++k) v += RED[tid * 16 + k];
        RED[272 + tid] = v;
      }
      __syncthreads();
      float rsum = RED[272 + m];
      for (int k = 0; k < 64; ++k) {
        float pr = __expf((float)lsrc[k] - rmax) / rsum;
        int c = chunk * 64 + k;
        float cs = (float)CSTAR[m * 1024 + c];
        CSTAR[m * 1024 + c] = (bf16)(pr * cs);
      }
    }
    __syncthreads();

    // ---- Phase E: att2 hidden = relu(attended @ w1^T + b1) -> HID1 ----
    {
      v8f acc[4] = {};
      int nt0 = wave * 4;
      auto lA = [&](int kt) { return load_a_bf16(CSTAR, 1024, kt * 32, lane); };
      gemm4(acc, lA, wp + OFF_ATT2W1, 32, nt0, 32, lane);
#pragma unroll
      for (int i = 0; i < 4; ++i)
        store_tile(HID1, 512, (nt0 + i) * 16, acc[i],
                   A2b1[(nt0 + i) * 16 + ncol], 1, mrow, ncol);
    }
    __syncthreads();

    // ---- Phase F: cHat = tanh(HID1 @ w2^T + b2) -> CHAT (X region) ----
    {
      v8f acc[4] = {};
      int nt0 = wave * 4;
      auto lA = [&](int kt) { return load_a_bf16(HID1, 512, kt * 32, lane); };
      gemm4(acc, lA, wp + OFF_ATT2W2, 32, nt0, 16, lane);
#pragma unroll
      for (int i = 0; i < 4; ++i)
        store_tile(CHAT, 512, (nt0 + i) * 16, acc[i],
                   A2b2[(nt0 + i) * 16 + ncol], 2, mrow, ncol);
    }
    __syncthreads();

    // ---- Phase G: g1/g2 hiddens over both=[attended(bf16), mem(f32)], K=1536 ----
    {
      int u = wave >> 2;  // waves 0-3 -> g1, 4-7 -> g2 (wave-uniform)
      const bf16* wp1 = wp + (u ? OFF_G2W1 : OFF_G1W1);
      const float* b1p = u ? G2b1 : G1b1;
      bf16* dsth = u ? HID1 : HID0;
      for (int p = 0; p < 2; ++p) {
        v8f acc[4] = {};
        int nt0 = (wave & 3) * 8 + p * 4;
        auto lA = [&](int kt) {
          return (kt < 32) ? load_a_bf16(CSTAR, 1024, kt * 32, lane)
                           : load_a_f32(MEMF, 512, (kt - 32) * 32, lane);
        };
        gemm4(acc, lA, wp1, 32, nt0, 48, lane);
#pragma unroll
        for (int i = 0; i < 4; ++i)
          store_tile(dsth, 512, (nt0 + i) * 16, acc[i],
                     b1p[(nt0 + i) * 16 + ncol], 1, mrow, ncol);
      }
    }
    __syncthreads();

    // ---- Phase H: gamma1/gamma2 + mem update ----
    {
      int nt0 = wave * 4;
      v8f acc1[4] = {}, acc2[4] = {};
      {
        auto lA = [&](int kt) { return load_a_bf16(HID0, 512, kt * 32, lane); };
        gemm4(acc1, lA, wp + OFF_G1W2, 32, nt0, 16, lane);
      }
      {
        auto lA = [&](int kt) { return load_a_bf16(HID1, 512, kt * 32, lane); };
        gemm4(acc2, lA, wp + OFF_G2W2, 32, nt0, 16, lane);
      }
#pragma unroll
      for (int i = 0; i < 4; ++i) {
        int c = (nt0 + i) * 16 + ncol;
        float b1v = G1b2[c], b2v = G2b2[c];
#pragma unroll
        for (int r = 0; r < 8; ++r) {
          int m = mrow + r;
          float g1v = sigmoid_f(acc1[i][r] + b1v);
          float g2v = sigmoid_f(acc2[i][r] + b2v);
          float mo = MEMF[m * 512 + c];
          float ch = (float)CHAT[m * 512 + c];
          MEMF[m * 512 + c] = g1v * mo + g2v * ch;
        }
      }
    }
    __syncthreads();
  }  // timestep loop

  // ---- Output MLP: hidden = relu([h | mem] @ out_w1^T + b1) ----
  {
    v8f acc[4] = {};
    int nt0 = wave * 4;
    auto lA = [&](int kt) {
      return (kt < 16) ? load_a_bf16(H, 512, kt * 32, lane)
                       : load_a_f32(MEMF, 512, (kt - 16) * 32, lane);
    };
    gemm4(acc, lA, wpack + OFF_OUTW1, 32, nt0, 32, lane);
#pragma unroll
    for (int i = 0; i < 4; ++i)
      store_tile(HID0, 512, (nt0 + i) * 16, acc[i],
                 out_b1[(nt0 + i) * 16 + ncol], 1, mrow, ncol);
  }
  __syncthreads();
  // final dot: y[m] = hidden[m] . out_w2 + out_b2
  {
    int m = tid >> 4, chunk = tid & 15;
    float s = 0.f;
    for (int k = 0; k < 32; ++k) {
      int c = chunk * 32 + k;
      s += (float)HID0[m * 512 + c] * out_w2[c];
    }
    float* S = (float*)HID1;
    S[m * 16 + chunk] = s;
    __syncthreads();
    if (tid < 16) {
      float acc = out_b2[0];
      for (int k = 0; k < 16; ++k) acc += S[tid * 16 + k];
      out[row0 + tid] = acc;
    }
  }
}

// ---------------- host ----------------
extern "C" void kernel_launch(void* const* d_in, const int* in_sizes, int n_in,
                              void* d_out, int out_size, void* d_ws, size_t ws_size,
                              hipStream_t stream) {
  (void)in_sizes; (void)n_in; (void)out_size; (void)ws_size;
  const int*   x_idx = (const int*)d_in[0];
  const float* x_av  = (const float*)d_in[1];
  const float* c_l0  = (const float*)d_in[2];
  const float* c_a0  = (const float*)d_in[3];
  const float* c_v0  = (const float*)d_in[4];
  const float* mem0  = (const float*)d_in[5];
  const float* embed = (const float*)d_in[6];
  bf16* wpack = (bf16*)d_ws;

  // pack all GEMM weights into bf16 WMMA B-fragment layout (deterministic, every call)
  struct M { int src; long off; int dout, din, dinP; };
  const M mats[15] = {
    {7,  OFF_WIH_L,  1024, 300,  320},
    {8,  OFF_WHH_L,  1024, 256,  256},
    {11, OFF_WIH_A,  512,  128,  128},
    {12, OFF_WHH_A,  512,  128,  128},
    {15, OFF_WIH_V,  512,  128,  128},
    {16, OFF_WHH_V,  512,  128,  128},
    {19, OFF_ATT1W1, 512,  1024, 1024},
    {21, OFF_ATT1W2, 1024, 512,  512},
    {23, OFF_ATT2W1, 512,  1024, 1024},
    {25, OFF_ATT2W2, 512,  512,  512},
    {27, OFF_G1W1,   512,  1536, 1536},
    {29, OFF_G1W2,   512,  512,  512},
    {31, OFF_G2W1,   512,  1536, 1536},
    {33, OFF_G2W2,   512,  512,  512},
    {35, OFF_OUTW1,  512,  1024, 1024},
  };
  for (int i = 0; i < 15; ++i) {
    pack_weights<<<256, THREADS, 0, stream>>>(
        (const float*)d_in[mats[i].src], wpack + mats[i].off,
        mats[i].dout, mats[i].din, mats[i].dinP);
  }

  mfn_main<<<BATCH / ROWS, THREADS, SMEM_BYTES, stream>>>(
      x_idx, x_av, c_l0, c_a0, c_v0, mem0, embed,
      (const float*)d_in[9],  (const float*)d_in[10],  // bih_l, bhh_l
      (const float*)d_in[13], (const float*)d_in[14],  // bih_a, bhh_a
      (const float*)d_in[17], (const float*)d_in[18],  // bih_v, bhh_v
      (const float*)d_in[20], (const float*)d_in[22],  // att1_b1, att1_b2
      (const float*)d_in[24], (const float*)d_in[26],  // att2_b1, att2_b2
      (const float*)d_in[28], (const float*)d_in[30],  // g1_b1, g1_b2
      (const float*)d_in[32], (const float*)d_in[34],  // g2_b1, g2_b2
      (const float*)d_in[36], (const float*)d_in[37],  // out_b1, out_w2
      (const float*)d_in[38],                          // out_b2
      wpack, (float*)d_out);
}
